// ContrastiveLoss_33569464385991
// MI455X (gfx1250) — compile-verified, matrix operand order checked by reading, not run on hardware
//
#include <hip/hip_runtime.h>
#include <hip/hip_bf16.h>

typedef __attribute__((ext_vector_type(16))) __bf16 v16bf;
typedef __attribute__((ext_vector_type(8)))  __bf16 v8bf;
typedef __attribute__((ext_vector_type(8)))  float  v8f;
typedef __attribute__((ext_vector_type(4)))  unsigned int v4u;
typedef __attribute__((ext_vector_type(8)))  int v8i;
typedef __attribute__((ext_vector_type(4)))  int v4i;

#define N_ROWS 8192
#define DIM    128
#define NCLS   43
#define MARGIN 1.0f

// LDS tile pitch: 128 bf16 (256B) + 8 bf16 (16B) TDM-inserted pad -> 272B.
// 272B = 68 DWORDs, so row r starts at bank (r*4)%64: the 16 rows of a
// fragment read hit 16 disjoint 4-bank groups -> conflict-free ds_load_b128.
#define LDS_PITCH      136
#define TILE_LDS_ELEMS (128 * LDS_PITCH)

// ---------------------------------------------------------------------------
// Workspace layout (d_ws):
//   [0 .. 15]              : float accum[4] = {pos_sum, neg_sum, pos_cnt, neg_cnt}
//   [256 .. 256+32KB)      : sq[N_ROWS]  (fp32 exact row norms)
//   [256+32KB .. +2MB)     : xb[N_ROWS*DIM]  (bf16 copy of x)
// ---------------------------------------------------------------------------

// Kernel 1: label histogram -> analytic pair counts; zero the sum accumulators.
__global__ void count_kernel(const int* __restrict__ y, float* __restrict__ accum) {
    __shared__ int hist[NCLS];
    const int t = threadIdx.x;
    if (t < NCLS) hist[t] = 0;
    __syncthreads();
    for (int i = t; i < N_ROWS; i += blockDim.x)
        atomicAdd(&hist[y[i]], 1);
    __syncthreads();
    if (t == 0) {
        long long ss = 0;
        for (int c = 0; c < NCLS; ++c) { long long n = hist[c]; ss += n * n; }
        accum[0] = 0.0f;                                        // pos_sum
        accum[1] = 0.0f;                                        // neg_sum
        accum[2] = (float)(ss - (long long)N_ROWS);             // pos pairs (i!=j, same label)
        accum[3] = (float)((long long)N_ROWS * N_ROWS - ss);    // neg pairs
    }
}

// Kernel 2: one wave32 per row: exact fp32 sum-of-squares + bf16 conversion.
__global__ __launch_bounds__(256)
void prep_kernel(const float* __restrict__ x, float* __restrict__ sq,
                 __bf16* __restrict__ xb) {
    const int wave = (int)((blockIdx.x * blockDim.x + threadIdx.x) >> 5);
    const int lane = threadIdx.x & 31;
    if (wave >= N_ROWS) return;

    const float4 v = reinterpret_cast<const float4*>(x + (size_t)wave * DIM)[lane];
    float s = v.x * v.x + v.y * v.y + v.z * v.z + v.w * v.w;

    __bf16 b4[4] = { (__bf16)v.x, (__bf16)v.y, (__bf16)v.z, (__bf16)v.w };
    *reinterpret_cast<uint2*>(xb + (size_t)wave * DIM + lane * 4) =
        *reinterpret_cast<uint2*>(b4);

    #pragma unroll
    for (int m = 16; m >= 1; m >>= 1) s += __shfl_xor(s, m, 32);
    if (lane == 0) sq[wave] = s;
}

// TDM: DMA a 128-row x 128-col bf16 tile (row-major, row stride 128) into LDS,
// inserting 16B of pad after every 256B row (pad_interval=64 DWORDs -> code 5,
// pad_amount=4 DWORDs -> code 3) to produce the 272B LDS pitch.
__device__ __forceinline__ void tdm_load_tile_128x128(const __bf16* gsrc,
                                                      unsigned lds_off) {
    const unsigned long long ga = (unsigned long long)gsrc;
    v4u g0;
    g0[0] = 1u;                                                  // count=1 (valid user D#)
    g0[1] = lds_off;                                             // lds_addr (bytes)
    g0[2] = (unsigned)ga;                                        // global_addr[31:0]
    g0[3] = (unsigned)((ga >> 32) & 0x01FFFFFFu) | (2u << 30);   // global_addr[56:32] | type=2
    v8i g1;
    g1[0] = (1 << 16)        // data_size = 2 bytes
          | (1 << 20)        // pad_enable
          | (5 << 22)        // pad_interval: 64 DWORDs (one 128-elem row)
          | (3 << 25);       // pad_amount: 4 DWORDs (16B)
    g1[1] = (int)((DIM & 0xFFFF) << 16);       // tensor_dim0 = 128 (lo16)
    g1[2] = (int)((N_ROWS & 0xFFFF) << 16);    // tensor_dim0 hi=0 | tensor_dim1 = 8192 (lo16)
    g1[3] = (int)(DIM << 16);                  // tensor_dim1 hi=0 | tile_dim0 = 128
    g1[4] = 128;                               // tile_dim1 = 128, tile_dim2 = 0
    g1[5] = DIM;                               // tensor_dim0_stride = 128 (lo32)
    g1[6] = 0;                                 // stride hi | tensor_dim1_stride lo
    g1[7] = 0;
    v4i gz = {0, 0, 0, 0};
#if __has_include(<hip/amd_detail/amd_gfx1250_TDM.h>)
    v8i gz8 = {0, 0, 0, 0, 0, 0, 0, 0};
    __builtin_amdgcn_tensor_load_to_lds(g0, g1, gz, gz, gz8, 0);
#else
    __builtin_amdgcn_tensor_load_to_lds(g0, g1, gz, gz, 0);
#endif
}

// Kernel 3: fused bf16-WMMA GEMM (x @ x^T) + loss epilogue + reduction.
// 128x128 output tile per 256-thread block; A/B tiles staged in LDS by the TDM.
// 8 waves in a 2(m) x 4(n) grid; each wave owns 64x32 = 4x2 subtiles of 16x16.
__global__ __launch_bounds__(256)
void gemm_loss_kernel(const __bf16* __restrict__ xb,
                      const float* __restrict__ sq,
                      const int*   __restrict__ y,
                      float* __restrict__ accum) {
    __shared__ __bf16 As[TILE_LDS_ELEMS];
    __shared__ __bf16 Bs[TILE_LDS_ELEMS];

    const int lane = threadIdx.x & 31;
    const int w    = threadIdx.x >> 5;
    const int wm   = w >> 2;        // 0..1
    const int wn   = w & 3;         // 0..3
    const int l16  = lane & 15;
    const int lh   = lane >> 4;     // 0/1: which K-half this lane holds

    const int mblk = blockIdx.y * 128;
    const int nblk = blockIdx.x * 128;

    // Wave 0 issues both tile DMAs and drains TENSORcnt; barrier releases all.
    if (w == 0) {
        tdm_load_tile_128x128(xb + (size_t)mblk * DIM,
                              (unsigned)(unsigned long long)(void*)As);
        tdm_load_tile_128x128(xb + (size_t)nblk * DIM,
                              (unsigned)(unsigned long long)(void*)Bs);
        __builtin_amdgcn_s_wait_tensorcnt(0);
    }
    __syncthreads();

    const int mb = wm * 64;   // row offset inside A tile
    const int nb = wn * 32;   // row offset inside B tile

    const v8f vzero = {0.f, 0.f, 0.f, 0.f, 0.f, 0.f, 0.f, 0.f};
    v8f acc[4][2];
    #pragma unroll
    for (int a = 0; a < 4; ++a)
        #pragma unroll
        for (int b = 0; b < 2; ++b) acc[a][b] = vzero;

    union Frag { v16bf v; v8bf h[2]; };

    #pragma unroll
    for (int kk = 0; kk < DIM; kk += 32) {
        // 16-bit A/B fragment layout (ISA 7.12.2): lane = row%16;
        // lanes 0-15 hold K = kk+0..7 and kk+16..23, lanes 16-31 hold +8.
        const int k0 = kk + lh * 8;

        Frag A[4];
        #pragma unroll
        for (int sm = 0; sm < 4; ++sm) {
            const __bf16* p = As + (mb + sm * 16 + l16) * LDS_PITCH + k0;
            A[sm].h[0] = *reinterpret_cast<const v8bf*>(p);
            A[sm].h[1] = *reinterpret_cast<const v8bf*>(p + 16);
        }
        Frag B[2];
        #pragma unroll
        for (int sn = 0; sn < 2; ++sn) {
            const __bf16* p = Bs + (nb + sn * 16 + l16) * LDS_PITCH + k0;
            B[sn].h[0] = *reinterpret_cast<const v8bf*>(p);
            B[sn].h[1] = *reinterpret_cast<const v8bf*>(p + 16);
        }

        #pragma unroll
        for (int sm = 0; sm < 4; ++sm)
            #pragma unroll
            for (int sn = 0; sn < 2; ++sn)
                acc[sm][sn] = __builtin_amdgcn_wmma_f32_16x16x32_bf16(
                    /*neg_a=*/false, A[sm].v, /*neg_b=*/false, B[sn].v,
                    /*c_mod=*/(short)0, acc[sm][sn],
                    /*reuse_a=*/false, /*reuse_b=*/false);
    }

    // Epilogue: C layout (ISA 7.12.2): element (vgpr r, lane) = C[M][N],
    // N = lane%16, M = r + 8*(lane>>4).
    float pos = 0.f, neg = 0.f;
    #pragma unroll
    for (int sn = 0; sn < 2; ++sn) {
        const int   j   = nblk + nb + sn * 16 + l16;
        const float sqj = sq[j];
        const int   yj  = y[j];
        #pragma unroll
        for (int sm = 0; sm < 4; ++sm) {
            const int ib = mblk + mb + sm * 16 + lh * 8;
            #pragma unroll
            for (int r = 0; r < 8; ++r) {
                const int   i = ib + r;
                const float d = sq[i] + sqj - 2.0f * acc[sm][sn][r];
                if (y[i] == yj) {
                    if (i != j) pos += d;
                } else {
                    neg += fmaxf(MARGIN - d, 0.0f);
                }
            }
        }
    }

    #pragma unroll
    for (int m = 16; m >= 1; m >>= 1) {
        pos += __shfl_xor(pos, m, 32);
        neg += __shfl_xor(neg, m, 32);
    }
    if (lane == 0) {
        atomicAdd(&accum[0], pos);
        atomicAdd(&accum[1], neg);
    }
}

// Kernel 4: combine into the scalar loss.
__global__ void final_kernel(const float* __restrict__ accum, float* __restrict__ out) {
    out[0] = 0.5f * (accum[0] / accum[2] + accum[1] / accum[3]);
}

extern "C" void kernel_launch(void* const* d_in, const int* in_sizes, int n_in,
                              void* d_out, int out_size, void* d_ws, size_t ws_size,
                              hipStream_t stream) {
    const float* x = (const float*)d_in[0];
    const int*   y = (const int*)d_in[1];
    float* out = (float*)d_out;

    char*   ws    = (char*)d_ws;
    float*  accum = (float*)ws;                                   // 4 floats
    float*  sq    = (float*)(ws + 256);                           // 8192 floats
    __bf16* xb    = (__bf16*)(ws + 256 + N_ROWS * sizeof(float)); // 8192*128 bf16

    count_kernel<<<1, 256, 0, stream>>>(y, accum);
    prep_kernel<<<N_ROWS / 8, 256, 0, stream>>>(x, sq, xb);
    gemm_loss_kernel<<<dim3(64, 64), 256, 0, stream>>>(xb, sq, y, accum);
    final_kernel<<<1, 1, 0, stream>>>(accum, out);
}
